// UnifiedMultiLevel_new_22368189677924
// MI455X (gfx1250) — matrix operand order repaired and checked by reading, not compiled, not run on hardware
//
#include <hip/hip_runtime.h>
#include <hip/hip_bf16.h>

// ---------------------------------------------------------------------------
// MI455X (gfx1250) implementation.
//  - decompose_kernel / reconstruct_kernel: per-(b,c)-row lifting in LDS
//  - pack_weights_kernel: gate weights -> bf16, WMMA-fragment-major layout
//  - gate_kernel: fused conv(k=3)+GELU+1x1+sigmoid+gating with
//    v_wmma_f32_16x16x32_bf16; operands are pure b128 loads (no per-element
//    conversion VALU in the K-loop)
// ---------------------------------------------------------------------------

#define TPB 256

constexpr int Cc   = 32;
constexpr int Ll   = 4096;
constexpr int Bb   = 64;
constexpr int CCH  = 128;   // 4*C stacked channels
constexpr int HCH  = 256;   // hidden channels of gate net
constexpr int NT   = 64;    // L-positions per gate workgroup
constexpr int XS   = 68;    // fp32 X tile stride (floats)
constexpr int XSB  = 144;   // bf16 X^T tile stride (ushorts, 288B = 9*32)
constexpr int HSB  = 272;   // bf16 h^T tile stride (ushorts, 544B = 17*32)

constexpr int NFRAG1 = 3 * 4 * 16;  // k-tap * ic-tile * m-tile = 192 fragments
constexpr int NFRAG2 = 8 * 8;       // k-tile * m-tile       =  64 fragments

typedef __attribute__((ext_vector_type(16))) __bf16 v16bf;
typedef __attribute__((ext_vector_type(8)))  float  v8f;

union Frag {
  unsigned short u[16];
  unsigned int   d[8];
  uint4          q[2];
  v16bf          v;
};

__device__ __forceinline__ int refl(int i, int n) {
  i = (i < 0) ? -i : i;
  return (i >= n) ? (2 * n - 2 - i) : i;
}
__device__ __forceinline__ float gelu_f(float x) {
  return 0.5f * x * (1.0f + erff(x * 0.70710678118654752f));
}
__device__ __forceinline__ float sigmoid_f(float x) {
  return 1.0f / (1.0f + expf(-x));
}
__device__ __forceinline__ unsigned int f2bf(float f) {
  unsigned int u = __float_as_uint(f);
  u += 0x7FFFu + ((u >> 16) & 1u);      // round-to-nearest-even
  return u >> 16;
}

// ---------------------------------------------------------------------------
// Kernel A: 3-level lifting decomposition, one workgroup per (b,c) row.
// ---------------------------------------------------------------------------
__global__ __launch_bounds__(TPB) void decompose_kernel(
    const float* __restrict__ x,
    const float* __restrict__ dec_P_w, const float* __restrict__ dec_P_b,
    const float* __restrict__ dec_U_w, const float* __restrict__ dec_U_b,
    const float* __restrict__ dec_aa_w, const float* __restrict__ dec_aa_b,
    float* __restrict__ coeffs)
{
  __shared__ float sA[Ll];
  __shared__ float sT1[Ll];
  __shared__ float sT2[Ll];

  const int bc = blockIdx.x;
  const int b  = bc / Cc;
  const int c  = bc % Cc;
  const int t  = threadIdx.x;

  const float* xr = x + (size_t)(b * Cc + c) * Ll;
  float* co = coeffs + (size_t)b * CCH * Ll;

  for (int j = t; j < Ll; j += TPB) sA[j] = xr[j];
  __syncthreads();

  for (int lvl = 0; lvl < 3; ++lvl) {
    const int d = 1 << lvl;
    float aaw[3], pw[5], uw[5];
    const float aab = dec_aa_b[lvl * Cc + c];
    const float pb  = dec_P_b[lvl * Cc + c];
    const float ub  = dec_U_b[lvl * Cc + c];
    for (int k = 0; k < 3; ++k) aaw[k] = dec_aa_w[(lvl * Cc + c) * 3 + k];
    for (int k = 0; k < 5; ++k) { pw[k] = dec_P_w[(lvl * Cc + c) * 5 + k];
                                  uw[k] = dec_U_w[(lvl * Cc + c) * 5 + k]; }

    // T1 = conv3(reflect(approx)) + b
    for (int j = t; j < Ll; j += TPB) {
      float s = aab;
      #pragma unroll
      for (int k = 0; k < 3; ++k) s += aaw[k] * sA[refl(j - 1 + k, Ll)];
      sT1[j] = s;
    }
    __syncthreads();
    // T2 = avgpool3(T1), zero-padded window, /3
    for (int j = t; j < Ll; j += TPB) {
      float s = sT1[j];
      if (j > 0)      s += sT1[j - 1];
      if (j < Ll - 1) s += sT1[j + 1];
      sT2[j] = s * (1.0f / 3.0f);
    }
    __syncthreads();
    // detail = T2 - gelu(conv5_d(reflect(T2)))  -> T1
    for (int j = t; j < Ll; j += TPB) {
      float s = pb;
      #pragma unroll
      for (int k = 0; k < 5; ++k) s += pw[k] * sT2[refl(j + d * (k - 2), Ll)];
      sT1[j] = sT2[j] - gelu_f(s);
    }
    __syncthreads();
    for (int j = t; j < Ll; j += TPB)
      co[(size_t)(lvl * Cc + c) * Ll + j] = sT1[j];
    // approx = 0.5*(T2 + gelu(conv5_d(reflect(detail)))) -> sA
    for (int j = t; j < Ll; j += TPB) {
      float s = ub;
      #pragma unroll
      for (int k = 0; k < 5; ++k) s += uw[k] * sT1[refl(j + d * (k - 2), Ll)];
      sA[j] = 0.5f * (sT2[j] + gelu_f(s));
    }
    __syncthreads();
  }
  for (int j = t; j < Ll; j += TPB)
    co[(size_t)(3 * Cc + c) * Ll + j] = sA[j];
}

// ---------------------------------------------------------------------------
// Pack gate weights to bf16 in WMMA-A-fragment-major order:
//   pw1[frag][lane][h], frag = (k*4 + ict)*16 + mtile, 512 halves per frag
//   pw2[frag][lane][h], frag = kt*8 + mtile2
// K order chosen here for A and B alike (contiguous): K(h) = h + (hi<<4)
// ---------------------------------------------------------------------------
__global__ __launch_bounds__(TPB) void pack_weights_kernel(
    const float* __restrict__ w1, const float* __restrict__ w2,
    unsigned short* __restrict__ pw1, unsigned short* __restrict__ pw2)
{
  const int gid  = blockIdx.x * TPB + threadIdx.x;
  const int lane = gid & 31;
  const int col  = lane & 15;
  const int hi   = lane >> 4;
  Frag f;
  if (gid < NFRAG1 * 32) {
    const int fr  = gid >> 5;
    const int k   = fr >> 6;         // 0..2
    const int ict = (fr >> 4) & 3;   // 0..3
    const int mt  = fr & 15;         // 0..15
    const int oc  = mt * 16 + col;
    const float* wrow = w1 + (size_t)oc * (CCH * 3) + k;
    #pragma unroll
    for (int h = 0; h < 16; ++h) {
      const int ic = ict * 32 + h + (hi << 4);
      f.u[h] = (unsigned short)f2bf(wrow[ic * 3]);
    }
    uint4* dst = (uint4*)pw1 + (size_t)fr * 64 + lane * 2;
    dst[0] = f.q[0]; dst[1] = f.q[1];
  } else if (gid < (NFRAG1 + NFRAG2) * 32) {
    const int fr  = (gid - NFRAG1 * 32) >> 5;
    const int kt  = fr >> 3;         // 0..7
    const int mt2 = fr & 7;          // 0..7
    const int oc2 = mt2 * 16 + col;
    const float* wrow = w2 + (size_t)oc2 * HCH;
    #pragma unroll
    for (int h = 0; h < 16; ++h) {
      const int oc = kt * 32 + h + (hi << 4);
      f.u[h] = (unsigned short)f2bf(wrow[oc]);
    }
    uint4* dst = (uint4*)pw2 + (size_t)fr * 64 + lane * 2;
    dst[0] = f.q[0]; dst[1] = f.q[1];
  }
}

// ---------------------------------------------------------------------------
// Kernel B: fused gate network with WMMA bf16 (fp32 accumulate).
//   h = gelu(W1 (*k=3) X + b1)   [M=256, K=3x128, N=64 per WG]
//   g = sigmoid(W2 h + b2)       [M=128, K=256,   N=64]
//   gated = X * g
// ---------------------------------------------------------------------------
__global__ __launch_bounds__(TPB) void gate_kernel(
    const float* __restrict__ coeffs,
    const unsigned short* __restrict__ pw1, const float* __restrict__ b1,
    const unsigned short* __restrict__ pw2, const float* __restrict__ b2,
    float* __restrict__ gated)
{
  __shared__ float          Xt [CCH * XS];         // fp32 [ch][pos]  (gating)
  __shared__ unsigned short XbT[(NT + 2) * XSB];   // bf16 [pos][ic]  (GEMM1 B)
  __shared__ unsigned short HtT[NT * HSB];         // bf16 [pos][oc]  (GEMM2 B)

  const int l0   = blockIdx.x * NT;
  const int b    = blockIdx.y;
  const int t    = threadIdx.x;
  const int wave = t >> 5;
  const int lane = t & 31;
  const int col  = lane & 15;   // N (or M-row) selector
  const int hi   = lane >> 4;   // 0/1 half-wave selector

  const float* xb = coeffs + (size_t)b * CCH * Ll;

  // Load X tile with reflect halo (positions l0-1 .. l0+64); bf16 copy once.
  for (int idx = t; idx < CCH * (NT + 2); idx += TPB) {
    const int ch = idx / (NT + 2);
    const int p  = idx - ch * (NT + 2);
    const float v = xb[(size_t)ch * Ll + refl(l0 - 1 + p, Ll)];
    Xt [ch * XS + p]  = v;
    XbT[p * XSB + ch] = (unsigned short)f2bf(v);
  }
  __syncthreads();

  // ---------------- GEMM1 ----------------
  v8f acc[2][4];
  #pragma unroll
  for (int i = 0; i < 2; ++i)
    #pragma unroll
    for (int n = 0; n < 4; ++n)
      #pragma unroll
      for (int r = 0; r < 8; ++r) acc[i][n][r] = 0.0f;

  const int mg = wave;  // this wave owns output rows [mg*32, mg*32+32)
  for (int k = 0; k < 3; ++k) {          // conv tap
    for (int ict = 0; ict < 4; ++ict) {  // K=128 in chunks of 32
      Frag a[2];
      #pragma unroll
      for (int i = 0; i < 2; ++i) {
        const int fr = (k * 4 + ict) * 16 + (mg * 2 + i);
        const uint4* ap = (const uint4*)pw1 + (size_t)fr * 64 + lane * 2;
        a[i].q[0] = ap[0]; a[i].q[1] = ap[1];
      }
      #pragma unroll
      for (int n = 0; n < 4; ++n) {
        Frag bfr;
        const uint4* bp =
            (const uint4*)&XbT[(n * 16 + col + k) * XSB + ict * 32 + (hi << 4)];
        bfr.q[0] = bp[0]; bfr.q[1] = bp[1];
        acc[0][n] = __builtin_amdgcn_wmma_f32_16x16x32_bf16(
            false, a[0].v, false, bfr.v, (short)0, acc[0][n], false, false);
        acc[1][n] = __builtin_amdgcn_wmma_f32_16x16x32_bf16(
            false, a[1].v, false, bfr.v, (short)0, acc[1][n], false, false);
      }
    }
  }
  // epilogue 1: bias + GELU -> HtT (bf16 pairs, b32 stores)
  #pragma unroll
  for (int i = 0; i < 2; ++i) {
    const int oc0 = (mg * 2 + i) * 16 + (hi << 3);
    #pragma unroll
    for (int n = 0; n < 4; ++n) {
      const int pos = n * 16 + col;
      unsigned int* hrow = (unsigned int*)&HtT[pos * HSB + oc0];
      #pragma unroll
      for (int r2 = 0; r2 < 4; ++r2) {
        const unsigned int lo =
            f2bf(gelu_f(acc[i][n][2 * r2]     + b1[oc0 + 2 * r2]));
        const unsigned int hh =
            f2bf(gelu_f(acc[i][n][2 * r2 + 1] + b1[oc0 + 2 * r2 + 1]));
        hrow[r2] = lo | (hh << 16);
      }
    }
  }
  __syncthreads();

  // ---------------- GEMM2 ----------------
  v8f acc2[4];
  #pragma unroll
  for (int n = 0; n < 4; ++n)
    #pragma unroll
    for (int r = 0; r < 8; ++r) acc2[n][r] = 0.0f;

  const int mt2 = wave;  // oc2 tile (128/16 = 8 tiles, one per wave)
  for (int kt = 0; kt < 8; ++kt) {
    Frag a;
    const uint4* ap = (const uint4*)pw2 + (size_t)(kt * 8 + mt2) * 64 + lane * 2;
    a.q[0] = ap[0]; a.q[1] = ap[1];
    #pragma unroll
    for (int n = 0; n < 4; ++n) {
      Frag bfr;
      const uint4* bp =
          (const uint4*)&HtT[(n * 16 + col) * HSB + kt * 32 + (hi << 4)];
      bfr.q[0] = bp[0]; bfr.q[1] = bp[1];
      acc2[n] = __builtin_amdgcn_wmma_f32_16x16x32_bf16(
          false, a.v, false, bfr.v, (short)0, acc2[n], false, false);
    }
  }
  // epilogue 2: gated = X * sigmoid(acc + b2)
  float* gb = gated + (size_t)b * CCH * Ll;
  #pragma unroll
  for (int n = 0; n < 4; ++n) {
    const int pos = n * 16 + col;
    #pragma unroll
    for (int r = 0; r < 8; ++r) {
      const int oc2 = mt2 * 16 + r + (hi << 3);
      const float gv = sigmoid_f(acc2[n][r] + b2[oc2]);
      const float xv = Xt[oc2 * XS + pos + 1];  // +1: skip left halo
      gb[(size_t)oc2 * Ll + l0 + pos] = xv * gv;
    }
  }
}

// ---------------------------------------------------------------------------
// Kernel C: 3-level lifting reconstruction, one workgroup per (b,c) row.
// ---------------------------------------------------------------------------
__global__ __launch_bounds__(TPB) void reconstruct_kernel(
    const float* __restrict__ gated,
    const float* __restrict__ rec_P_w, const float* __restrict__ rec_P_b,
    const float* __restrict__ rec_U_w, const float* __restrict__ rec_U_b,
    float* __restrict__ out)
{
  __shared__ float sCur[Ll];
  __shared__ float sDet[Ll];
  __shared__ float sX[Ll];

  const int bc = blockIdx.x;
  const int b  = bc / Cc;
  const int c  = bc % Cc;
  const int t  = threadIdx.x;

  const float* g = gated + (size_t)b * CCH * Ll;

  for (int j = t; j < Ll; j += TPB) sCur[j] = g[(size_t)(3 * Cc + c) * Ll + j];

  for (int idx = 0; idx < 3; ++idx) {
    const int d    = 4 >> idx;       // 4, 2, 1
    const int dlvl = 2 - idx;        // detail level 2, 1, 0
    float upw[5], ppw[5];
    const float upb = rec_U_b[idx * Cc + c];
    const float ppb = rec_P_b[idx * Cc + c];
    for (int k = 0; k < 5; ++k) { upw[k] = rec_U_w[(idx * Cc + c) * 5 + k];
                                  ppw[k] = rec_P_w[(idx * Cc + c) * 5 + k]; }

    for (int j = t; j < Ll; j += TPB)
      sDet[j] = g[(size_t)(dlvl * Cc + c) * Ll + j];
    __syncthreads();
    // x_rec = 2*current - gelu(conv5_d(reflect(detail)))
    for (int j = t; j < Ll; j += TPB) {
      float s = upb;
      #pragma unroll
      for (int k = 0; k < 5; ++k) s += upw[k] * sDet[refl(j + d * (k - 2), Ll)];
      sX[j] = 2.0f * sCur[j] - gelu_f(s);
    }
    __syncthreads();
    // current = x_rec + gelu(conv5_d(reflect(x_rec)))
    for (int j = t; j < Ll; j += TPB) {
      float s = ppb;
      #pragma unroll
      for (int k = 0; k < 5; ++k) s += ppw[k] * sX[refl(j + d * (k - 2), Ll)];
      sCur[j] = sX[j] + gelu_f(s);
    }
    __syncthreads();
  }
  for (int j = t; j < Ll; j += TPB)
    out[(size_t)(b * Cc + c) * Ll + j] = sCur[j];
}

// ---------------------------------------------------------------------------
extern "C" void kernel_launch(void* const* d_in, const int* in_sizes, int n_in,
                              void* d_out, int out_size, void* d_ws, size_t ws_size,
                              hipStream_t stream) {
  const float* x        = (const float*)d_in[0];
  const float* dec_P_w  = (const float*)d_in[1];
  const float* dec_P_b  = (const float*)d_in[2];
  const float* dec_U_w  = (const float*)d_in[3];
  const float* dec_U_b  = (const float*)d_in[4];
  const float* dec_aa_w = (const float*)d_in[5];
  const float* dec_aa_b = (const float*)d_in[6];
  const float* rec_P_w  = (const float*)d_in[7];
  const float* rec_P_b  = (const float*)d_in[8];
  const float* rec_U_w  = (const float*)d_in[9];
  const float* rec_U_b  = (const float*)d_in[10];
  const float* gate_w1  = (const float*)d_in[11];
  const float* gate_b1  = (const float*)d_in[12];
  const float* gate_w2  = (const float*)d_in[13];
  const float* gate_b2  = (const float*)d_in[14];
  float* out = (float*)d_out;

  // workspace layout: packed bf16 weights | coeffs | gated
  unsigned short* pw1 = (unsigned short*)d_ws;         // 98304 halves
  unsigned short* pw2 = pw1 + (size_t)NFRAG1 * 512;    // 32768 halves
  float* coeffs = (float*)((char*)d_ws + 262144);      // B*128*L fp32
  float* gated  = coeffs + (size_t)Bb * CCH * Ll;      // B*128*L fp32

  pack_weights_kernel<<<(NFRAG1 + NFRAG2) * 32 / TPB, TPB, 0, stream>>>(
      gate_w1, gate_w2, pw1, pw2);

  decompose_kernel<<<Bb * Cc, TPB, 0, stream>>>(
      x, dec_P_w, dec_P_b, dec_U_w, dec_U_b, dec_aa_w, dec_aa_b, coeffs);

  dim3 gg(Ll / NT, Bb);
  gate_kernel<<<gg, TPB, 0, stream>>>(
      coeffs, pw1, gate_b1, pw2, gate_b2, gated);

  reconstruct_kernel<<<Bb * Cc, TPB, 0, stream>>>(
      gated, rec_P_w, rec_P_b, rec_U_w, rec_U_b, out);
}
